// MultiheadAttention_20650202759513
// MI455X (gfx1250) — compile-verified, matrix operand order checked by reading, not run on hardware
//
#include <hip/hip_runtime.h>
#include <hip/hip_bf16.h>
#include <math.h>

// ---------------------------------------------------------------------------
// Multi-head attention (T=2048, B=8, F=768, H=3, dh=256) for gfx1250 / MI455X
//   Stage 0: one-pass fp32 -> bf16 conversion of x and the 4 weight matrices
//   Stage 1: fused Q/K/V projection  (bf16 WMMA, fp32 accumulate, 32x64 tiles)
//   Stage 2: flash attention, one wave per 16-query x head tile
//   Stage 3: output projection (32x64 tiles)
// ---------------------------------------------------------------------------

typedef __bf16 bf16_t;
typedef __attribute__((ext_vector_type(16))) __bf16 v16bf;
typedef __attribute__((ext_vector_type(8)))  __bf16 v8bf;
typedef __attribute__((ext_vector_type(8)))  float  v8f;

#define T_DIM 2048
#define B_DIM 8
#define F_DIM 768
#define H_DIM 3
#define DH    256
#define M_DIM (T_DIM * B_DIM)
#define INV_SCALE 0.125f   // 1/sqrt(768/12) = 1/8

// ---- WMMA fragment loader (bf16 source) ------------------------------------
// 16-bit A-matrix 16x32 layout (ISA 7.12.2): lane L holds row M=L%16;
// lanes 0-15 hold K = k0+{0..7} (elems 0-7) and k0+16+{0..7} (elems 8-15);
// lanes 16-31 the same shifted by +8 in K. A B-operand (32x16, column n fed
// from a contiguous-K source row n) uses the identical pattern.
static __device__ __forceinline__ v16bf frag_from_bf16(const bf16_t* __restrict__ src,
                                                       int row0, int ld, int k0) {
    const int lane = threadIdx.x & 31;
    const bf16_t* p = src + (long)(row0 + (lane & 15)) * ld + k0 + ((lane >> 4) << 3);
    v16bf f;
#pragma unroll
    for (int i = 0; i < 8; ++i) {
        f[i]     = p[i];        // 8 contiguous bf16 = one 16B load
        f[i + 8] = p[i + 16];   // second 16B load, 32B further
    }
    return f;
}

static __device__ __forceinline__ v8f wmma_bf16(v16bf a, v16bf b, v8f c) {
    return __builtin_amdgcn_wmma_f32_16x16x32_bf16(false, a, false, b,
                                                   (short)0, c, false, false);
}

// ---- Stage 0: bulk fp32 -> bf16 conversion ---------------------------------
__global__ __launch_bounds__(256)
void cvt_f32_bf16_kernel(const float* __restrict__ src, bf16_t* __restrict__ dst,
                         int n) {
    const int i = (blockIdx.x * 256 + threadIdx.x) * 8;
    if (i + 8 <= n) {
        v8bf o;
#pragma unroll
        for (int e = 0; e < 8; ++e) o[e] = (__bf16)src[i + e];
        *(v8bf*)(dst + i) = o;
    }
}

// ---- Stage 1: fused QKV projection ----------------------------------------
// grid: (M/32, F/64, 3). One wave computes a 32x64 strip (8 WMMAs / k-step).
// q,k stored as (b*H+h, T, dh) bf16; v stored transposed as (b*H+h, dh, T).
__global__ __launch_bounds__(32)
void qkv_proj_kernel(const bf16_t* __restrict__ xb,
                     const bf16_t* __restrict__ Wqb,
                     const bf16_t* __restrict__ Wkb,
                     const bf16_t* __restrict__ Wvb,
                     const float* __restrict__ bq,
                     const float* __restrict__ bk,
                     const float* __restrict__ bv,
                     bf16_t* __restrict__ qbuf, bf16_t* __restrict__ kbuf,
                     bf16_t* __restrict__ vT) {
    const int m0    = blockIdx.x * 32;
    const int n0    = blockIdx.y * 64;
    const int which = blockIdx.z;

    const bf16_t* W    = (which == 0) ? Wqb : (which == 1) ? Wkb : Wvb;
    const float*  bias = (which == 0) ? bq  : (which == 1) ? bk  : bv;

    v8f acc[2][4] = {};
    for (int k0 = 0; k0 < F_DIM; k0 += 32) {
        v16bf a0 = frag_from_bf16(xb, m0,      F_DIM, k0);
        v16bf a1 = frag_from_bf16(xb, m0 + 16, F_DIM, k0);
#pragma unroll
        for (int nt = 0; nt < 4; ++nt) {
            v16bf bfr = frag_from_bf16(W, n0 + nt * 16, F_DIM, k0); // B[k][n]=W[n][k]
            acc[0][nt] = wmma_bf16(a0, bfr, acc[0][nt]);
            acc[1][nt] = wmma_bf16(a1, bfr, acc[1][nt]);
        }
    }

    const int lane  = threadIdx.x & 31;
    const int col   = lane & 15;
    const int rbase = (lane >> 4) << 3;           // rows j (lanes 0-15) / j+8
#pragma unroll
    for (int mt = 0; mt < 2; ++mt) {
#pragma unroll
        for (int nt = 0; nt < 4; ++nt) {
            const int   n    = n0 + nt * 16 + col;
            const float bval = bias[n];
            const int   h    = n / DH;
            const int   d    = n % DH;
#pragma unroll
            for (int j = 0; j < 8; ++j) {
                const int m = m0 + mt * 16 + rbase + j;   // m = t*B + b
                const int t = m / B_DIM;
                const int b = m % B_DIM;
                const float val = acc[mt][nt][j] + bval;
                const long  bh  = (long)(b * H_DIM + h);
                if (which == 2) {
                    vT[(bh * DH + d) * T_DIM + t] = (__bf16)val;
                } else {
                    bf16_t* dst = (which == 0) ? qbuf : kbuf;
                    dst[(bh * T_DIM + t) * DH + d] = (__bf16)val;
                }
            }
        }
    }
}

// ---- Stage 2: flash attention ----------------------------------------------
// grid: (T/16, B*H). One wave owns 16 query rows of one head, streams key/value
// chunks of 32, keeps online-softmax state and a 16x256 f32 accumulator.
__global__ __launch_bounds__(32)
void flash_attn_kernel(const bf16_t* __restrict__ qbuf,
                       const bf16_t* __restrict__ kbuf,
                       const bf16_t* __restrict__ vT,
                       bf16_t* __restrict__ obuf) {
    __shared__ __bf16 plds[16 * 32];              // P tile bounce buffer

    const int t0 = blockIdx.x * 16;
    const int bh = blockIdx.y;
    const int b  = bh / H_DIM;
    const int h  = bh % H_DIM;

    const bf16_t* Q = qbuf + (long)bh * T_DIM * DH;
    const bf16_t* K = kbuf + (long)bh * T_DIM * DH;
    const bf16_t* V = vT   + (long)bh * DH * T_DIM;

    const int lane  = threadIdx.x & 31;
    const int col   = lane & 15;
    const int rbase = (lane >> 4) << 3;

    // Resident Q fragments: dh=256 -> 8 K-steps of 32
    v16bf qf[8];
#pragma unroll
    for (int kk = 0; kk < 8; ++kk) qf[kk] = frag_from_bf16(Q, t0, DH, kk * 32);

    v8f oacc[16];
    {
        v8f z = {};
#pragma unroll
        for (int dt = 0; dt < 16; ++dt) oacc[dt] = z;
    }
    float mrow[8], lrow[8];
#pragma unroll
    for (int j = 0; j < 8; ++j) { mrow[j] = -1e30f; lrow[j] = 0.0f; }

    for (int s0 = 0; s0 < T_DIM; s0 += 32) {
        // hint-prefetch the next key chunk (global_prefetch_b8)
        if (s0 + 32 < T_DIM)
            __builtin_prefetch(K + (long)(s0 + 32) * DH + lane * 256, 0, 1);

        // ---- scores: 16 x 32 (two 16x16 tiles), K-dim = dh = 256
        v8f sc0 = {}, sc1 = {};
#pragma unroll
        for (int kk = 0; kk < 8; ++kk) {
            v16bf kf0 = frag_from_bf16(K, s0,      DH, kk * 32);  // B[d][s]=k[s][d]
            v16bf kf1 = frag_from_bf16(K, s0 + 16, DH, kk * 32);
            sc0 = wmma_bf16(qf[kk], kf0, sc0);
            sc1 = wmma_bf16(qf[kk], kf1, sc1);
        }

        // ---- online softmax over the 32 new scores, per row
#pragma unroll
        for (int j = 0; j < 8; ++j) {
            float a0 = sc0[j] * INV_SCALE;
            float a1 = sc1[j] * INV_SCALE;
            float tmax = fmaxf(a0, a1);
#pragma unroll
            for (int msk = 8; msk; msk >>= 1)
                tmax = fmaxf(tmax, __shfl_xor(tmax, msk, 32));     // within 16-lane half
            const float newm = fmaxf(mrow[j], tmax);
            const float corr = __expf(mrow[j] - newm);
            const float p0   = __expf(a0 - newm);
            const float p1   = __expf(a1 - newm);
            float psum = p0 + p1;
#pragma unroll
            for (int msk = 8; msk; msk >>= 1)
                psum += __shfl_xor(psum, msk, 32);
            lrow[j] = lrow[j] * corr + psum;
            mrow[j] = newm;
#pragma unroll
            for (int dt = 0; dt < 16; ++dt) oacc[dt][j] *= corr;

            const int prow = rbase + j;                 // C-layout -> LDS [row][col]
            plds[prow * 32 + col]      = (__bf16)p0;
            plds[prow * 32 + 16 + col] = (__bf16)p1;
        }
        __syncthreads();

        // ---- re-read P in A-fragment layout, then P(16x32) x V(32x256)
        v16bf pa = frag_from_bf16((const bf16_t*)plds, 0, 32, 0);
#pragma unroll
        for (int dt = 0; dt < 16; ++dt) {
            v16bf vf = frag_from_bf16(V, dt * 16, T_DIM, s0);      // B[s][d]=vT[d][s]
            oacc[dt] = wmma_bf16(pa, vf, oacc[dt]);
        }
        __syncthreads();
    }

    // ---- normalize and store to (T*B, F) bf16
#pragma unroll
    for (int j = 0; j < 8; ++j) {
        const float inv = 1.0f / lrow[j];
        const int   t   = t0 + rbase + j;
#pragma unroll
        for (int dt = 0; dt < 16; ++dt) {
            const long off = ((long)t * B_DIM + b) * F_DIM + h * DH + dt * 16 + col;
            obuf[off] = (__bf16)(oacc[dt][j] * inv);
        }
    }
}

// ---- Stage 3: output projection (32x64 tiles) ------------------------------
__global__ __launch_bounds__(32)
void out_proj_kernel(const bf16_t* __restrict__ obuf,
                     const bf16_t* __restrict__ Wob, const float* __restrict__ bo,
                     float* __restrict__ out) {
    const int m0 = blockIdx.x * 32;
    const int n0 = blockIdx.y * 64;

    v8f acc[2][4] = {};
    for (int k0 = 0; k0 < F_DIM; k0 += 32) {
        v16bf a0 = frag_from_bf16(obuf, m0,      F_DIM, k0);
        v16bf a1 = frag_from_bf16(obuf, m0 + 16, F_DIM, k0);
#pragma unroll
        for (int nt = 0; nt < 4; ++nt) {
            v16bf bfr = frag_from_bf16(Wob, n0 + nt * 16, F_DIM, k0);
            acc[0][nt] = wmma_bf16(a0, bfr, acc[0][nt]);
            acc[1][nt] = wmma_bf16(a1, bfr, acc[1][nt]);
        }
    }

    const int lane  = threadIdx.x & 31;
    const int col   = lane & 15;
    const int rbase = (lane >> 4) << 3;
#pragma unroll
    for (int mt = 0; mt < 2; ++mt) {
#pragma unroll
        for (int nt = 0; nt < 4; ++nt) {
            const int   n    = n0 + nt * 16 + col;
            const float bval = bo[n];
#pragma unroll
            for (int j = 0; j < 8; ++j) {
                const int m = m0 + mt * 16 + rbase + j;
                out[(long)m * F_DIM + n] = acc[mt][nt][j] + bval;
            }
        }
    }
}

// ---------------------------------------------------------------------------
extern "C" void kernel_launch(void* const* d_in, const int* in_sizes, int n_in,
                              void* d_out, int out_size, void* d_ws, size_t ws_size,
                              hipStream_t stream) {
    const float* x  = (const float*)d_in[0];
    const float* Wq = (const float*)d_in[1];
    const float* bq = (const float*)d_in[2];
    const float* Wk = (const float*)d_in[3];
    const float* bk = (const float*)d_in[4];
    const float* Wv = (const float*)d_in[5];
    const float* bv = (const float*)d_in[6];
    const float* Wo = (const float*)d_in[7];
    const float* bo = (const float*)d_in[8];

    const size_t qkv_elems = (size_t)B_DIM * H_DIM * T_DIM * DH;   // 12.58M (== M*F)
    const size_t w_elems   = (size_t)F_DIM * F_DIM;                // 589824

    bf16_t* qbuf = (bf16_t*)d_ws;
    bf16_t* kbuf = qbuf + qkv_elems;
    bf16_t* vT   = kbuf + qkv_elems;
    bf16_t* obuf = vT   + qkv_elems;
    bf16_t* xb   = obuf + qkv_elems;
    bf16_t* Wqb  = xb   + qkv_elems;
    bf16_t* Wkb  = Wqb  + w_elems;
    bf16_t* Wvb  = Wkb  + w_elems;
    bf16_t* Wob  = Wvb  + w_elems;

    // Stage 0: convert activations + weights to bf16 (single streaming pass)
    {
        const int nx = (int)(M_DIM * (size_t)F_DIM);
        cvt_f32_bf16_kernel<<<dim3((nx / 8 + 255) / 256), 256, 0, stream>>>(x, xb, nx);
        const int nw = (int)w_elems;
        dim3 gw((nw / 8 + 255) / 256);
        cvt_f32_bf16_kernel<<<gw, 256, 0, stream>>>(Wq, Wqb, nw);
        cvt_f32_bf16_kernel<<<gw, 256, 0, stream>>>(Wk, Wkb, nw);
        cvt_f32_bf16_kernel<<<gw, 256, 0, stream>>>(Wv, Wvb, nw);
        cvt_f32_bf16_kernel<<<gw, 256, 0, stream>>>(Wo, Wob, nw);
    }

    // Stage 1: Q/K/V projections
    dim3 g1(M_DIM / 32, F_DIM / 64, 3);
    qkv_proj_kernel<<<g1, 32, 0, stream>>>(xb, Wqb, Wkb, Wvb, bq, bk, bv,
                                           qbuf, kbuf, vT);

    // Stage 2: flash attention
    dim3 g2(T_DIM / 16, B_DIM * H_DIM);
    flash_attn_kernel<<<g2, 32, 0, stream>>>(qbuf, kbuf, vT, obuf);

    // Stage 3: output projection
    dim3 g3(M_DIM / 32, F_DIM / 64);
    out_proj_kernel<<<g3, 32, 0, stream>>>(obuf, Wob, bo, (float*)d_out);
}
